// MultiHeadSelfAttention_91328184582525
// MI455X (gfx1250) — matrix operand order repaired and checked by reading, not compiled, not run on hardware
//
#include <hip/hip_runtime.h>
#include <hip/hip_bf16.h>
#include <stdint.h>

#define S_LEN   4096
#define D_MODEL 1024
#define NQKV    3072
#define NHEADS  16
#define HDIM    64

typedef __bf16 bf16_t;
typedef bf16_t bf16x16 __attribute__((ext_vector_type(16)));
typedef float  v8f     __attribute__((ext_vector_type(8)));

union Frag {
    bf16x16 bf;
    uint4   q[2];
    unsigned short u[16];
};

static __device__ __forceinline__ unsigned short f32_to_bf16_bits(float f) {
    union { float f; unsigned int u; } c; c.f = f;
    unsigned int r = 0x7FFFu + ((c.u >> 16) & 1u);
    return (unsigned short)((c.u + r) >> 16);
}

// A-frag (16x32 bf16): lane holds row = lane&15, khalf = lane>>4.
// elems 0..7  -> K = kbase + khalf*8 + i        (contiguous 8)
// elems 8..15 -> K = kbase + 16 + khalf*8 + i-8 (contiguous 8)
static __device__ __forceinline__ void load_a_frag(Frag& f, const unsigned short* base,
                                                   size_t row, int ld, int kbase, int khalf) {
    const unsigned short* p = base + row * (size_t)ld + kbase + khalf * 8;
    f.q[0] = *(const uint4*)p;
    f.q[1] = *(const uint4*)(p + 16);
}

// B-frag (32x16 bf16): lane holds col n = lane&15, khalf = lane>>4.
// elems 0..7  -> K = kbase + khalf*8 + i        (contiguous 8)
// elems 8..15 -> K = kbase + 16 + khalf*8 + i-8 (contiguous 8)
static __device__ __forceinline__ void load_b_frag(Frag& f, const unsigned short* rowptr,
                                                   int kbase, int khalf) {
    const unsigned short* p = rowptr + kbase + khalf * 8;
    f.q[0] = *(const uint4*)p;
    f.q[1] = *(const uint4*)(p + 16);
}

static __device__ __forceinline__ v8f wmma_bf16(const Frag& a, const Frag& b, v8f c) {
    return __builtin_amdgcn_wmma_f32_16x16x32_bf16(
        false, a.bf, false, b.bf, (short)0, c, false, false);
}

// ---------------------------------------------------------------- converts
__global__ void cvt_f32_to_bf16(const float* __restrict__ src,
                                unsigned short* __restrict__ dst, int n) {
    int i = blockIdx.x * blockDim.x + threadIdx.x;
    int stride = gridDim.x * blockDim.x;
    for (; i < n; i += stride) dst[i] = f32_to_bf16_bits(src[i]);
}

// ---------------------------------------------------------------- QKV GEMM
// Per wave: 32(M) x 64(N) tile -> 8 accumulators, B fragments reused across
// two A rows (1.5 b128 loads per WMMA).
// C[m][n] = sum_k x[m][k] * w_qkv[n][k] + b_qkv[n]; scatter to Q/K/V^T bf16.
__global__ void __launch_bounds__(256)
qkv_gemm(const unsigned short* __restrict__ xb,
         const unsigned short* __restrict__ wb,
         const float* __restrict__ bias,
         unsigned short* __restrict__ qout,   // [H][S][64]
         unsigned short* __restrict__ kout,   // [H][S][64]
         unsigned short* __restrict__ vtout)  // [H][64][S]
{
    const int lane   = threadIdx.x & 31;
    const int wave   = threadIdx.x >> 5;
    const int khalf  = lane >> 4;
    const int lane16 = lane & 15;
    const int m0     = blockIdx.x * 32;
    const int nbase  = blockIdx.y * 512 + wave * 64;

    v8f acc[8];
#pragma unroll
    for (int i = 0; i < 8; ++i) acc[i] = (v8f){};

    for (int k0 = 0; k0 < D_MODEL; k0 += 32) {
        Frag a0, a1;
        load_a_frag(a0, xb, (size_t)(m0 + lane16),      D_MODEL, k0, khalf);
        load_a_frag(a1, xb, (size_t)(m0 + 16 + lane16), D_MODEL, k0, khalf);
#pragma unroll
        for (int j = 0; j < 4; ++j) {
            Frag b;
            load_b_frag(b, wb + (size_t)(nbase + 16 * j + lane16) * D_MODEL, k0, khalf);
            acc[j]     = wmma_bf16(a0, b, acc[j]);
            acc[4 + j] = wmma_bf16(a1, b, acc[4 + j]);
        }
    }

    const float QSCALE = 0.18033688011112042f; // (1/sqrt(64)) * log2(e)
#pragma unroll
    for (int i = 0; i < 2; ++i) {
#pragma unroll
        for (int j = 0; j < 4; ++j) {
            v8f a = acc[i * 4 + j];
            const int n  = nbase + j * 16 + lane16;
            const float bv = bias[n];
            const int which = n >> 10;
            const int d = n & 1023;
            const int h = d >> 6, hd = d & 63;
            const int mbase = m0 + i * 16 + khalf * 8;
            if (which == 2) {
                union { unsigned short us[8]; uint4 v; } pk;
#pragma unroll
                for (int r = 0; r < 8; ++r) pk.us[r] = f32_to_bf16_bits(a[r] + bv);
                *(uint4*)(vtout + ((size_t)h * 64 + hd) * S_LEN + mbase) = pk.v;
            } else {
                const float scale = (which == 0) ? QSCALE : 1.0f;
                unsigned short* dst =
                    (which == 0 ? qout : kout) + (size_t)h * S_LEN * HDIM + hd;
#pragma unroll
                for (int r = 0; r < 8; ++r) {
                    dst[(size_t)(mbase + r) * HDIM] =
                        f32_to_bf16_bits((a[r] + bv) * scale);
                }
            }
        }
    }
}

// ---------------------------------------------------------------- flash attention
// One wave per 16-row query tile; keys streamed in tiles of 64 (4 score tiles,
// 16 WMMAs per iteration) so the softmax shuffle/rescale VALU cost per key is
// halved vs 32-key tiles. Online softmax in base-2 domain; P bounced via LDS
// to convert D-layout -> A-layout.
__global__ void __launch_bounds__(256)
attn_kernel(const unsigned short* __restrict__ qb,   // [H][S][64], pre-scaled
            const unsigned short* __restrict__ kb,   // [H][S][64]
            const unsigned short* __restrict__ vtb,  // [H][64][S]
            unsigned short* __restrict__ ctx)        // [S][D] bf16
{
    __shared__ __align__(16) unsigned short lds_p[8][16 * 64]; // 2KB per wave

    const int lane   = threadIdx.x & 31;
    const int wave   = threadIdx.x >> 5;
    const int khalf  = lane >> 4;
    const int lane16 = lane & 15;
    const int h      = blockIdx.y;
    const int m0     = (blockIdx.x * 8 + wave) * 16;

    const unsigned short* qh = qb  + (size_t)h * S_LEN * HDIM;
    const unsigned short* kh = kb  + (size_t)h * S_LEN * HDIM;
    const unsigned short* vh = vtb + (size_t)h * HDIM * S_LEN;

    Frag qa0, qa1;
    load_a_frag(qa0, qh, (size_t)(m0 + lane16), HDIM, 0,  khalf);
    load_a_frag(qa1, qh, (size_t)(m0 + lane16), HDIM, 32, khalf);

    v8f acc[4];
#pragma unroll
    for (int j = 0; j < 4; ++j) acc[j] = (v8f){};
    float mrow[8], lrow[8];
#pragma unroll
    for (int r = 0; r < 8; ++r) { mrow[r] = -1e30f; lrow[r] = 0.0f; }

    unsigned short* pl = &lds_p[wave][0];

    for (int k0 = 0; k0 < S_LEN; k0 += 64) {
        // ---- scores: four 16x16 tiles covering keys k0 .. k0+63
        v8f s[4];
#pragma unroll
        for (int t = 0; t < 4; ++t) {
            s[t] = (v8f){};
            Frag b;
            const unsigned short* krow = kh + (size_t)(k0 + 16 * t + lane16) * HDIM;
            load_b_frag(b, krow, 0,  khalf);
            s[t] = wmma_bf16(qa0, b, s[t]);
            load_b_frag(b, krow, 32, khalf);
            s[t] = wmma_bf16(qa1, b, s[t]);
        }

        // ---- online softmax (base-2); reductions stay within 16-lane halves
#pragma unroll
        for (int r = 0; r < 8; ++r) {
            float v = fmaxf(fmaxf(s[0][r], s[1][r]), fmaxf(s[2][r], s[3][r]));
            v = fmaxf(v, __shfl_xor(v, 1, 32));
            v = fmaxf(v, __shfl_xor(v, 2, 32));
            v = fmaxf(v, __shfl_xor(v, 4, 32));
            v = fmaxf(v, __shfl_xor(v, 8, 32));
            const float mnew = fmaxf(mrow[r], v);
            const float corr = exp2f(mrow[r] - mnew);
            const float p0 = exp2f(s[0][r] - mnew);
            const float p1 = exp2f(s[1][r] - mnew);
            const float p2 = exp2f(s[2][r] - mnew);
            const float p3 = exp2f(s[3][r] - mnew);
            float sum = (p0 + p1) + (p2 + p3);
            sum += __shfl_xor(sum, 1, 32);
            sum += __shfl_xor(sum, 2, 32);
            sum += __shfl_xor(sum, 4, 32);
            sum += __shfl_xor(sum, 8, 32);
            lrow[r] = lrow[r] * corr + sum;
            mrow[r] = mnew;
            acc[0][r] *= corr; acc[1][r] *= corr;
            acc[2][r] *= corr; acc[3][r] *= corr;
            // stage P (D-layout coords: row = r + 8*khalf, col = 16*t + lane16)
            const int mr = r + 8 * khalf;
            pl[mr * 64 +  0 + lane16] = f32_to_bf16_bits(p0);
            pl[mr * 64 + 16 + lane16] = f32_to_bf16_bits(p1);
            pl[mr * 64 + 32 + lane16] = f32_to_bf16_bits(p2);
            pl[mr * 64 + 48 + lane16] = f32_to_bf16_bits(p3);
        }

        // ---- read P back as two 16x32 A-fragments
        Frag pa0, pa1;
        {
            const unsigned short* p = pl + lane16 * 64 + khalf * 8;
            pa0.q[0] = *(const uint4*)p;
            pa0.q[1] = *(const uint4*)(p + 16);
            pa1.q[0] = *(const uint4*)(p + 32);
            pa1.q[1] = *(const uint4*)(p + 48);
        }

        // ---- ctx += P @ V  (V^T rows are contiguous in kk)
#pragma unroll
        for (int j = 0; j < 4; ++j) {
            Frag vb;
            const unsigned short* vrow = vh + (size_t)(16 * j + lane16) * S_LEN;
            load_b_frag(vb, vrow, k0,      khalf);
            acc[j] = wmma_bf16(pa0, vb, acc[j]);
            load_b_frag(vb, vrow, k0 + 32, khalf);
            acc[j] = wmma_bf16(pa1, vb, acc[j]);
        }
    }

    // ---- finalize: divide by row sums, store ctx bf16 [S][D]
#pragma unroll
    for (int j = 0; j < 4; ++j) {
        const int d = h * HDIM + j * 16 + lane16;
#pragma unroll
        for (int r = 0; r < 8; ++r) {
            const int s = m0 + khalf * 8 + r;
            ctx[(size_t)s * D_MODEL + d] = f32_to_bf16_bits(acc[j][r] / lrow[r]);
        }
    }
}

// ---------------------------------------------------------------- out projection
// Same 32x64-per-wave blocking as qkv_gemm; fp32 output with bias.
__global__ void __launch_bounds__(256)
out_gemm(const unsigned short* __restrict__ ctxb,
         const unsigned short* __restrict__ wob,
         const float* __restrict__ bias,
         float* __restrict__ out)
{
    const int lane   = threadIdx.x & 31;
    const int wave   = threadIdx.x >> 5;
    const int khalf  = lane >> 4;
    const int lane16 = lane & 15;
    const int m0     = blockIdx.x * 32;
    const int nbase  = blockIdx.y * 512 + wave * 64;

    v8f acc[8];
#pragma unroll
    for (int i = 0; i < 8; ++i) acc[i] = (v8f){};

    for (int k0 = 0; k0 < D_MODEL; k0 += 32) {
        Frag a0, a1;
        load_a_frag(a0, ctxb, (size_t)(m0 + lane16),      D_MODEL, k0, khalf);
        load_a_frag(a1, ctxb, (size_t)(m0 + 16 + lane16), D_MODEL, k0, khalf);
#pragma unroll
        for (int j = 0; j < 4; ++j) {
            Frag b;
            load_b_frag(b, wob + (size_t)(nbase + 16 * j + lane16) * D_MODEL, k0, khalf);
            acc[j]     = wmma_bf16(a0, b, acc[j]);
            acc[4 + j] = wmma_bf16(a1, b, acc[4 + j]);
        }
    }

#pragma unroll
    for (int i = 0; i < 2; ++i) {
#pragma unroll
        for (int j = 0; j < 4; ++j) {
            v8f a = acc[i * 4 + j];
            const int n = nbase + j * 16 + lane16;
            const float bv = bias[n];
            const int mbase = m0 + i * 16 + khalf * 8;
#pragma unroll
            for (int r = 0; r < 8; ++r) {
                out[(size_t)(mbase + r) * D_MODEL + n] = a[r] + bv;
            }
        }
    }
}

// ---------------------------------------------------------------- launch
extern "C" void kernel_launch(void* const* d_in, const int* in_sizes, int n_in,
                              void* d_out, int out_size, void* d_ws, size_t ws_size,
                              hipStream_t stream) {
    const float* x      = (const float*)d_in[0];
    const float* w_qkv  = (const float*)d_in[1];
    const float* b_qkv  = (const float*)d_in[2];
    const float* w_o    = (const float*)d_in[3];
    const float* b_o    = (const float*)d_in[4];
    float* out = (float*)d_out;

    char* ws = (char*)d_ws;
    unsigned short* xb    = (unsigned short*)(ws + ((size_t)0  << 20)); // 8 MB
    unsigned short* wqkvb = (unsigned short*)(ws + ((size_t)8  << 20)); // 6 MB
    unsigned short* wob   = (unsigned short*)(ws + ((size_t)14 << 20)); // 2 MB
    unsigned short* qb    = (unsigned short*)(ws + ((size_t)16 << 20)); // 8 MB
    unsigned short* kb    = (unsigned short*)(ws + ((size_t)24 << 20)); // 8 MB
    unsigned short* vtb   = (unsigned short*)(ws + ((size_t)32 << 20)); // 8 MB
    unsigned short* ctxb  = (unsigned short*)(ws + ((size_t)40 << 20)); // 8 MB

    cvt_f32_to_bf16<<<2048, 256, 0, stream>>>(x,     xb,    S_LEN * D_MODEL);
    cvt_f32_to_bf16<<<2048, 256, 0, stream>>>(w_qkv, wqkvb, NQKV  * D_MODEL);
    cvt_f32_to_bf16<<<1024, 256, 0, stream>>>(w_o,   wob,   D_MODEL * D_MODEL);

    qkv_gemm   <<<dim3(S_LEN / 32, NQKV / 512),    256, 0, stream>>>(xb, wqkvb, b_qkv, qb, kb, vtb);
    attn_kernel<<<dim3(S_LEN / 128, NHEADS),       256, 0, stream>>>(qb, kb, vtb, ctxb);
    out_gemm   <<<dim3(S_LEN / 32, D_MODEL / 512), 256, 0, stream>>>(ctxb, wob, b_o, out);
}
